// PointTransformerSeg_4793183502916
// MI455X (gfx1250) — matrix operand order repaired and checked by reading, not compiled, not run on hardware
//
#include <hip/hip_runtime.h>
#include <hip/hip_bf16.h>
#include <math.h>

#define BB   4
#define NP   2048
#define KNN  16
#define DM   512

typedef __attribute__((ext_vector_type(16))) _Float16 v16h;
typedef __attribute__((ext_vector_type(8)))  _Float16 v8h;
typedef __attribute__((ext_vector_type(8)))  float    v8f;

static inline int cdiv(int a, int b){ return (a + b - 1) / b; }

// ---------------------------------------------------------------------------
// WMMA GEMM:  C[M,N] = A[M,K] * W[K,N] (+bias) (+add) (relu)
// Block = 128 threads (4 waves) -> 64(M) x 64(N) macro-tile.
// Tiles staged through LDS as f16 (coalesced float4 global loads, b128 LDS
// reads); each wave owns 16 M-rows and 4 accumulators across 64 N-cols.
// Edge handling via address-clamp + select (no divergent loads).
// ---------------------------------------------------------------------------
__global__ __launch_bounds__(128) void k_gemm_wmma(
    const float* __restrict__ A, const float* __restrict__ W,
    const float* __restrict__ bias, const float* __restrict__ add,
    float* __restrict__ C, int M, int K, int N, int relu)
{
  __shared__ __align__(16) _Float16 As[64][40];   // [m][k]  (pad to 40 halves)
  __shared__ __align__(16) _Float16 Bt[64][40];   // [n][k]  (transposed)

  const int bn64 = blockIdx.x * 64;
  const int bm64 = blockIdx.y * 64;
  const int lane = threadIdx.x;                   // 0..31
  const int ty   = threadIdx.y;                   // 0..3 (wave id)
  const int tid  = ty * 32 + lane;                // 0..127
  const int hf   = lane >> 4;
  const int l16  = lane & 15;
  const bool fastK = ((K & 31) == 0);             // no k-guards needed

  v8f acc[4];
#pragma unroll
  for (int nt = 0; nt < 4; ++nt) acc[nt] = (v8f){};

  for (int kb = 0; kb < K; kb += 32) {
    // ---------------- stage A (64 x 32) ----------------
    if (fastK) {
      int r = tid >> 1, hh = tid & 1;
      int grow = bm64 + r;
      const float4* p = (const float4*)(A + (size_t)(grow < M ? grow : M - 1) * K + kb + hh * 16);
      float4 f0 = p[0], f1 = p[1], f2 = p[2], f3 = p[3];
      bool ok = grow < M;
      v8h s0, s1;
      s0[0]=(_Float16)(ok?f0.x:0.f); s0[1]=(_Float16)(ok?f0.y:0.f);
      s0[2]=(_Float16)(ok?f0.z:0.f); s0[3]=(_Float16)(ok?f0.w:0.f);
      s0[4]=(_Float16)(ok?f1.x:0.f); s0[5]=(_Float16)(ok?f1.y:0.f);
      s0[6]=(_Float16)(ok?f1.z:0.f); s0[7]=(_Float16)(ok?f1.w:0.f);
      s1[0]=(_Float16)(ok?f2.x:0.f); s1[1]=(_Float16)(ok?f2.y:0.f);
      s1[2]=(_Float16)(ok?f2.z:0.f); s1[3]=(_Float16)(ok?f2.w:0.f);
      s1[4]=(_Float16)(ok?f3.x:0.f); s1[5]=(_Float16)(ok?f3.y:0.f);
      s1[6]=(_Float16)(ok?f3.z:0.f); s1[7]=(_Float16)(ok?f3.w:0.f);
      *(v8h*)&As[r][hh * 16]     = s0;
      *(v8h*)&As[r][hh * 16 + 8] = s1;
    } else {
#pragma unroll 4
      for (int idx = tid; idx < 64 * 32; idx += 128) {
        int r = idx >> 5, kk = idx & 31;
        int grow = bm64 + r, gk = kb + kk;
        float v = A[(size_t)(grow < M ? grow : M - 1) * K + (gk < K ? gk : K - 1)];
        As[r][kk] = (_Float16)((grow < M && gk < K) ? v : 0.f);
      }
    }
    // ---------------- stage B (32 x 64, transposed into Bt) ----------------
    if (fastK) {
      int kr = tid >> 2, nc = (tid & 3) * 16;
      int gn0 = bn64 + nc;
      const float* wrow = W + (size_t)(kb + kr) * N;
      if (((N & 3) == 0) && (gn0 + 15 < N)) {
        const float4* p = (const float4*)(wrow + gn0);
        float4 f0 = p[0], f1 = p[1], f2 = p[2], f3 = p[3];
        Bt[nc+ 0][kr]=(_Float16)f0.x; Bt[nc+ 1][kr]=(_Float16)f0.y;
        Bt[nc+ 2][kr]=(_Float16)f0.z; Bt[nc+ 3][kr]=(_Float16)f0.w;
        Bt[nc+ 4][kr]=(_Float16)f1.x; Bt[nc+ 5][kr]=(_Float16)f1.y;
        Bt[nc+ 6][kr]=(_Float16)f1.z; Bt[nc+ 7][kr]=(_Float16)f1.w;
        Bt[nc+ 8][kr]=(_Float16)f2.x; Bt[nc+ 9][kr]=(_Float16)f2.y;
        Bt[nc+10][kr]=(_Float16)f2.z; Bt[nc+11][kr]=(_Float16)f2.w;
        Bt[nc+12][kr]=(_Float16)f3.x; Bt[nc+13][kr]=(_Float16)f3.y;
        Bt[nc+14][kr]=(_Float16)f3.z; Bt[nc+15][kr]=(_Float16)f3.w;
      } else {
#pragma unroll
        for (int j = 0; j < 16; ++j) {
          int gn = gn0 + j;
          float v = wrow[gn < N ? gn : (N - 1)];
          Bt[nc + j][kr] = (_Float16)(gn < N ? v : 0.f);
        }
      }
    } else {
#pragma unroll 4
      for (int idx = tid; idx < 32 * 64; idx += 128) {
        int kr = idx >> 6, nn = idx & 63;
        int gk = kb + kr, gn = bn64 + nn;
        float v = W[(size_t)(gk < K ? gk : K - 1) * N + (gn < N ? gn : N - 1)];
        Bt[nn][kr] = (_Float16)((gk < K && gn < N) ? v : 0.f);
      }
    }
    __syncthreads();

    // ---------------- fragments + WMMA ----------------
    int mloc = ty * 16 + l16;
    v8h a0 = *(const v8h*)&As[mloc][8 * hf];
    v8h a1 = *(const v8h*)&As[mloc][16 + 8 * hf];
    v16h af;
#pragma unroll
    for (int e = 0; e < 8; ++e) { af[e] = a0[e]; af[8 + e] = a1[e]; }
#pragma unroll
    for (int nt = 0; nt < 4; ++nt) {
      v8h b0 = *(const v8h*)&Bt[nt * 16 + l16][8 * hf];
      v8h b1 = *(const v8h*)&Bt[nt * 16 + l16][16 + 8 * hf];
      v16h bf;
#pragma unroll
      for (int e = 0; e < 8; ++e) { bf[e] = b0[e]; bf[8 + e] = b1[e]; }
      acc[nt] = __builtin_amdgcn_wmma_f32_16x16x32_f16(false, af, false, bf,
                                                       (short)0, acc[nt], false, false);
    }
    __syncthreads();
  }

  // ---------------- epilogue ----------------
#pragma unroll
  for (int nt = 0; nt < 4; ++nt) {
#pragma unroll
    for (int r = 0; r < 8; ++r) {
      int gm = bm64 + ty * 16 + r + 8 * hf;
      int gn = bn64 + nt * 16 + l16;
      if (gm < M && gn < N) {
        float v = acc[nt][r];
        if (bias) v += bias[gn];
        if (add)  v += add[(size_t)gm * N + gn];
        if (relu) v = v > 0.f ? v : 0.f;
        C[(size_t)gm * N + gn] = v;
      }
    }
  }
}

// ---------------------------------------------------------------------------
// xyz = x[..., :3]
// ---------------------------------------------------------------------------
__global__ void k_slice3(const float* __restrict__ x, float* __restrict__ xyz, int total)
{
  int i = blockIdx.x * blockDim.x + threadIdx.x;
  if (i >= total) return;
  int c = i % 3; int bn = i / 3;
  xyz[i] = x[(size_t)bn * 6 + c];
}

// ---------------------------------------------------------------------------
// kNN (k=16), stable insertion sort == stable top_k(-dist)
// ---------------------------------------------------------------------------
__global__ void k_knn16(const float* __restrict__ q, const float* __restrict__ r,
                        int Nq, int Nr, int* __restrict__ idx)
{
  int t = blockIdx.x * blockDim.x + threadIdx.x;
  if (t >= BB * Nq) return;
  int b = t / Nq;
  const float* qp = q + (size_t)t * 3;
  float qx = qp[0], qy = qp[1], qz = qp[2];
  const float* R = r + (size_t)b * Nr * 3;
  float bd[KNN]; int bi[KNN];
#pragma unroll
  for (int j = 0; j < KNN; ++j) { bd[j] = 3.0e38f; bi[j] = 0; }
  for (int j = 0; j < Nr; ++j) {
    float dx = R[j*3] - qx, dy = R[j*3+1] - qy, dz = R[j*3+2] - qz;
    float d = dx*dx + dy*dy + dz*dz;
    if (d < bd[KNN-1]) {
      int p = KNN - 1;
      while (p > 0 && bd[p-1] > d) { bd[p] = bd[p-1]; bi[p] = bi[p-1]; --p; }
      bd[p] = d; bi[p] = j;
    }
  }
#pragma unroll
  for (int j = 0; j < KNN; ++j) idx[(size_t)t * KNN + j] = bi[j];
}

// kNN (k=3) with normalized inverse-distance weights (transition_up)
__global__ void k_knn3w(const float* __restrict__ q, const float* __restrict__ r,
                        int Nq, int Nr, int* __restrict__ idx, float* __restrict__ wout)
{
  int t = blockIdx.x * blockDim.x + threadIdx.x;
  if (t >= BB * Nq) return;
  int b = t / Nq;
  const float* qp = q + (size_t)t * 3;
  float qx = qp[0], qy = qp[1], qz = qp[2];
  const float* R = r + (size_t)b * Nr * 3;
  float bd[3]; int bi[3];
#pragma unroll
  for (int j = 0; j < 3; ++j) { bd[j] = 3.0e38f; bi[j] = 0; }
  for (int j = 0; j < Nr; ++j) {
    float dx = R[j*3] - qx, dy = R[j*3+1] - qy, dz = R[j*3+2] - qz;
    float d = dx*dx + dy*dy + dz*dz;
    if (d < bd[2]) {
      int p = 2;
      while (p > 0 && bd[p-1] > d) { bd[p] = bd[p-1]; bi[p] = bi[p-1]; --p; }
      bd[p] = d; bi[p] = j;
    }
  }
  float w0 = 1.f / (bd[0] + 1e-8f);
  float w1 = 1.f / (bd[1] + 1e-8f);
  float w2 = 1.f / (bd[2] + 1e-8f);
  float s = w0 + w1 + w2;
  idx[t*3+0] = bi[0]; idx[t*3+1] = bi[1]; idx[t*3+2] = bi[2];
  wout[t*3+0] = w0 / s; wout[t*3+1] = w1 / s; wout[t*3+2] = w2 / s;
}

// ---------------------------------------------------------------------------
// out[b,j,c] = src[b, idx[b,j], c]
// ---------------------------------------------------------------------------
__global__ void k_gather(const float* __restrict__ src, const int* __restrict__ idx,
                         float* __restrict__ out, int L, int Ns, int Cc, long total)
{
  long i = (long)blockIdx.x * blockDim.x + threadIdx.x;
  if (i >= total) return;
  int  c  = (int)(i % Cc);
  long bl = i / Cc;
  int  b  = (int)(bl / L);
  int  id = idx[bl];
  out[i] = src[((size_t)b * Ns + id) * Cc + c];
}

// rel[b,n,k,:] = xyz[b,n,:] - xyz[b, idx[b,n,k], :]
__global__ void k_relpos(const float* __restrict__ xyz, const int* __restrict__ idx,
                         float* __restrict__ out, int N, long total)
{
  long i = (long)blockIdx.x * blockDim.x + threadIdx.x;
  if (i >= total) return;
  int  c  = (int)(i % 3);
  long r  = i / 3;
  long bn = r / KNN;
  int  b  = (int)(bn / N);
  int  id = idx[r];
  out[i] = xyz[(size_t)bn * 3 + c] - xyz[((size_t)b * N + id) * 3 + c];
}

// transition_down grouping: concat(xyz_gather - new_xyz, feats_gather)
__global__ void k_tdcat(const float* __restrict__ xyzr, const float* __restrict__ nxyz,
                        const float* __restrict__ feats, const int* __restrict__ idx,
                        float* __restrict__ out, int M, int Ns, int Cin, long total)
{
  long i = (long)blockIdx.x * blockDim.x + threadIdx.x;
  if (i >= total) return;
  int  Ct = Cin + 3;
  int  c  = (int)(i % Ct);
  long r  = i / Ct;
  long bm = r / KNN;
  int  b  = (int)(bm / M);
  int  id = idx[r];
  if (c < 3)
    out[i] = xyzr[((size_t)b * Ns + id) * 3 + c] - nxyz[(size_t)bm * 3 + c];
  else
    out[i] = feats[((size_t)b * Ns + id) * Cin + (c - 3)];
}

// t = q[:, :, None] - kf + pos
__global__ void k_qkp(const float* __restrict__ q, const float* __restrict__ kf,
                      const float* __restrict__ pos, float* __restrict__ out, long total)
{
  long i = (long)blockIdx.x * blockDim.x + threadIdx.x;
  if (i >= total) return;
  int  f  = (int)(i % DM);
  long r  = i / DM;
  long bn = r / KNN;
  out[i] = q[(size_t)bn * DM + f] - kf[i] + pos[i];
}

// softmax over the neighbor (k) axis, scaled by 1/sqrt(DM); in place
__global__ void k_softmaxk(float* __restrict__ a, long totalBNF)
{
  long i = (long)blockIdx.x * blockDim.x + threadIdx.x;
  if (i >= totalBNF) return;
  int  f    = (int)(i % DM);
  long bn   = i / DM;
  size_t base = (size_t)bn * KNN * DM + f;
  const float scale = 0.04419417382415922f;   // 1/sqrt(512)
  float m = -3.0e38f;
#pragma unroll
  for (int k = 0; k < KNN; ++k) {
    float v = a[base + (size_t)k * DM] * scale;
    m = v > m ? v : m;
  }
  float e[KNN]; float s = 0.f;
#pragma unroll
  for (int k = 0; k < KNN; ++k) {
    float v = __expf(a[base + (size_t)k * DM] * scale - m);
    e[k] = v; s += v;
  }
  float inv = 1.f / s;
#pragma unroll
  for (int k = 0; k < KNN; ++k) a[base + (size_t)k * DM] = e[k] * inv;
}

// res[b,n,f] = sum_k attn * (v + pos)
__global__ void k_attnred(const float* __restrict__ a, const float* __restrict__ v,
                          const float* __restrict__ pos, float* __restrict__ res, long totalBNF)
{
  long i = (long)blockIdx.x * blockDim.x + threadIdx.x;
  if (i >= totalBNF) return;
  int  f  = (int)(i % DM);
  long bn = i / DM;
  size_t base = (size_t)bn * KNN * DM + f;
  float s = 0.f;
#pragma unroll
  for (int k = 0; k < KNN; ++k) {
    size_t o = base + (size_t)k * DM;
    s += a[o] * (v[o] + pos[o]);
  }
  res[i] = s;
}

// ---------------------------------------------------------------------------
// Farthest point sampling: one block per batch, sequential scan
// ---------------------------------------------------------------------------
__global__ void k_fps(const float* __restrict__ xyz, int N, int npoint, int* __restrict__ out)
{
  __shared__ float dmin[2048];
  __shared__ float rv[256];
  __shared__ int   ri[256];
  __shared__ int   s_far;
  int b = blockIdx.x;
  const float* X = xyz + (size_t)b * N * 3;
  int tid = threadIdx.x;
  for (int j = tid; j < N; j += blockDim.x) dmin[j] = 1.0e10f;
  if (tid == 0) s_far = 0;
  __syncthreads();
  for (int t = 0; t < npoint; ++t) {
    int far = s_far;
    if (tid == 0) out[b * npoint + t] = far;
    float cx = X[far*3], cy = X[far*3+1], cz = X[far*3+2];
    float bv = -1.f; int bi = 0x7fffffff;
    for (int j = tid; j < N; j += blockDim.x) {
      float dx = X[j*3] - cx, dy = X[j*3+1] - cy, dz = X[j*3+2] - cz;
      float d = dx*dx + dy*dy + dz*dz;
      float dm = dmin[j];
      if (d < dm) { dm = d; dmin[j] = d; }
      if (dm > bv || (dm == bv && j < bi)) { bv = dm; bi = j; }
    }
    rv[tid] = bv; ri[tid] = bi;
    __syncthreads();
    for (int s = 128; s > 0; s >>= 1) {
      if (tid < s) {
        if (rv[tid+s] > rv[tid] || (rv[tid+s] == rv[tid] && ri[tid+s] < ri[tid])) {
          rv[tid] = rv[tid+s]; ri[tid] = ri[tid+s];
        }
      }
      __syncthreads();
    }
    if (tid == 0) s_far = ri[0];
    __syncthreads();
  }
}

// ---------------------------------------------------------------------------
// BatchNorm (population stats over all rows) + ReLU
// ---------------------------------------------------------------------------
__global__ void k_colstats(const float* __restrict__ X, int rows, int C,
                           float* __restrict__ mean, float* __restrict__ rstd)
{
  __shared__ float ss[256], sq[256];
  int c = blockIdx.x;
  float s = 0.f, q = 0.f;
  for (int r = threadIdx.x; r < rows; r += blockDim.x) {
    float v = X[(size_t)r * C + c];
    s += v; q += v * v;
  }
  ss[threadIdx.x] = s; sq[threadIdx.x] = q;
  __syncthreads();
  for (int st = 128; st > 0; st >>= 1) {
    if (threadIdx.x < st) { ss[threadIdx.x] += ss[threadIdx.x+st]; sq[threadIdx.x] += sq[threadIdx.x+st]; }
    __syncthreads();
  }
  if (threadIdx.x == 0) {
    float m = ss[0] / rows;
    float var = sq[0] / rows - m * m;
    if (var < 0.f) var = 0.f;
    mean[c] = m;
    rstd[c] = rsqrtf(var + 1e-5f);
  }
}

__global__ void k_bnrelu(float* __restrict__ X, const float* __restrict__ g,
                         const float* __restrict__ be, const float* __restrict__ mean,
                         const float* __restrict__ rstd, int Cc, long total)
{
  long i = (long)blockIdx.x * blockDim.x + threadIdx.x;
  if (i >= total) return;
  int c = (int)(i % Cc);
  float v = g[c] * (X[i] - mean[c]) * rstd[c] + be[c];
  X[i] = v > 0.f ? v : 0.f;
}

// max over k axis
__global__ void k_maxk(const float* __restrict__ gg, float* __restrict__ out, int Cc, long totalBMC)
{
  long i = (long)blockIdx.x * blockDim.x + threadIdx.x;
  if (i >= totalBMC) return;
  int  c  = (int)(i % Cc);
  long bm = i / Cc;
  size_t base = (size_t)bm * KNN * Cc + c;
  float m = -3.0e38f;
#pragma unroll
  for (int k = 0; k < KNN; ++k) {
    float v = gg[base + (size_t)k * Cc];
    m = v > m ? v : m;
  }
  out[i] = m;
}

// out = sum_j w3[j]*f1[idx3[j]] + f2
__global__ void k_interp3(const float* __restrict__ f1, const float* __restrict__ f2,
                          const int* __restrict__ i3, const float* __restrict__ w3,
                          float* __restrict__ out, int Nf, int Nc, int Cc, long total)
{
  long i = (long)blockIdx.x * blockDim.x + threadIdx.x;
  if (i >= total) return;
  int  c  = (int)(i % Cc);
  long bn = i / Cc;
  int  b  = (int)(bn / Nf);
  float s = 0.f;
#pragma unroll
  for (int j = 0; j < 3; ++j) {
    int id = i3[bn*3 + j];
    s += w3[bn*3 + j] * f1[((size_t)b * Nc + id) * Cc + c];
  }
  out[i] = s + f2[i];
}

// ===========================================================================
// Host side: param walker, arena, orchestration
// ===========================================================================
struct Lin { const float* w; const float* b; };
struct LBN { const float* w; const float* b; const float* g; const float* be; };
struct TBp { Lin fc1, fc2, d1, d2, g1, g2; const float *wq, *wk, *wv; };
struct Net {
  Lin fc1a, fc1b, fc2a, fc2b, fc2c, fc3a, fc3b, fc3c;
  TBp t1, t2, tf[4], tutf[4];
  LBN td0[4], td1[4], tuf1[4], tuf2[4];
};
struct Cur { void* const* d; int i; const float* nx() { return (const float*)d[i++]; } };

static void rdLin(Cur& c, bool s, Lin& L) {
  if (s) { L.b = c.nx(); L.w = c.nx(); } else { L.w = c.nx(); L.b = c.nx(); }
}
static void rdLBN(Cur& c, bool s, LBN& L) {
  if (s) { L.b = c.nx(); L.be = c.nx(); L.g = c.nx(); L.w = c.nx(); }
  else   { L.w = c.nx(); L.b = c.nx(); L.g = c.nx(); L.be = c.nx(); }
}
static void rdTB(Cur& c, bool s, TBp& t) {
  if (s) {
    rdLin(c,s,t.d1); rdLin(c,s,t.d2); rdLin(c,s,t.fc1); rdLin(c,s,t.fc2);
    rdLin(c,s,t.g1); rdLin(c,s,t.g2);
    t.wk = c.nx(); t.wq = c.nx(); t.wv = c.nx();
  } else {
    rdLin(c,s,t.fc1); rdLin(c,s,t.fc2);
    t.wq = c.nx(); t.wk = c.nx(); t.wv = c.nx();
    rdLin(c,s,t.d1); rdLin(c,s,t.d2); rdLin(c,s,t.g1); rdLin(c,s,t.g2);
  }
}
static void readNet(Cur& c, bool s, Net& P) {
  if (s) { // sorted-key pytree order
    rdLin(c,s,P.fc1a); rdLin(c,s,P.fc1b); rdLin(c,s,P.fc2a); rdLin(c,s,P.fc2b); rdLin(c,s,P.fc2c);
    rdLin(c,s,P.fc3a); rdLin(c,s,P.fc3b); rdLin(c,s,P.fc3c);
    rdTB(c,s,P.t1); rdTB(c,s,P.t2);
    for (int i = 0; i < 4; ++i) { rdLBN(c,s,P.td0[i]); rdLBN(c,s,P.td1[i]); }
    for (int i = 0; i < 4; ++i) rdTB(c,s,P.tf[i]);
    for (int i = 0; i < 4; ++i) { rdLBN(c,s,P.tuf1[i]); rdLBN(c,s,P.tuf2[i]); }
    for (int i = 0; i < 4; ++i) rdTB(c,s,P.tutf[i]);
  } else { // dict insertion order
    rdLin(c,s,P.fc1a); rdLin(c,s,P.fc1b); rdTB(c,s,P.t1);
    for (int i = 0; i < 4; ++i) { rdLBN(c,s,P.td0[i]); rdLBN(c,s,P.td1[i]); }
    for (int i = 0; i < 4; ++i) rdTB(c,s,P.tf[i]);
    for (int i = 0; i < 4; ++i) { rdLBN(c,s,P.tuf1[i]); rdLBN(c,s,P.tuf2[i]); }
    for (int i = 0; i < 4; ++i) rdTB(c,s,P.tutf[i]);
    rdLin(c,s,P.fc2a); rdLin(c,s,P.fc2b); rdLin(c,s,P.fc2c); rdTB(c,s,P.t2);
    rdLin(c,s,P.fc3a); rdLin(c,s,P.fc3b); rdLin(c,s,P.fc3c);
  }
}

struct WS {
  float *big0, *big1, *big2, *big3;
  float *x, *q, *kx, *vx, *res;
  float *hA, *hB, *gA, *gB, *rel;
  int   *idx, *fps, *i3;
  float *w3, *mean, *rstd;
  float *sxyz[5], *sh[5];
};

static void gemm(hipStream_t st, const float* A, const float* W, const float* bias,
                 const float* add, float* C, int M, int K, int N, bool relu)
{
  dim3 g((unsigned)cdiv(N, 64), (unsigned)cdiv(M, 64), 1);
  dim3 b(32, 4, 1);
  k_gemm_wmma<<<g, b, 0, st>>>(A, W, bias, add, C, M, K, N, relu ? 1 : 0);
}

static void bnrelu(hipStream_t st, float* X, const LBN& p, int rows, int C, WS& w)
{
  k_colstats<<<C, 256, 0, st>>>(X, rows, C, w.mean, w.rstd);
  long total = (long)rows * C;
  k_bnrelu<<<(int)((total + 255) / 256), 256, 0, st>>>(X, p.g, p.be, w.mean, w.rstd, C, total);
}

static void transformer(hipStream_t st, const TBp& t, const float* xyz, float* h,
                        int N, int din, WS& w)
{
  int  BN   = BB * N;
  long BNK  = (long)BN * KNN;
  long BNKF = BNK * DM;
  long BNF  = (long)BN * DM;
  k_knn16<<<cdiv(BN, 128), 128, 0, st>>>(xyz, xyz, N, N, w.idx);
  gemm(st, h,   t.fc1.w, t.fc1.b, nullptr, w.x,  BN, din, DM, false);
  gemm(st, w.x, t.wq,    nullptr, nullptr, w.q,  BN, DM,  DM, false);
  gemm(st, w.x, t.wk,    nullptr, nullptr, w.kx, BN, DM,  DM, false);
  gemm(st, w.x, t.wv,    nullptr, nullptr, w.vx, BN, DM,  DM, false);
  int ewb = (int)((BNKF + 255) / 256);
  k_gather<<<ewb, 256, 0, st>>>(w.kx, w.idx, w.big0, N * KNN, N, DM, BNKF);  // kf
  k_gather<<<ewb, 256, 0, st>>>(w.vx, w.idx, w.big1, N * KNN, N, DM, BNKF);  // v
  k_relpos<<<(int)((BNK * 3 + 255) / 256), 256, 0, st>>>(xyz, w.idx, w.rel, N, BNK * 3);
  gemm(st, w.rel,  t.d1.w, t.d1.b, nullptr, w.big3, (int)BNK, 3,  DM, true);   // relu(d1)
  gemm(st, w.big3, t.d2.w, t.d2.b, nullptr, w.big2, (int)BNK, DM, DM, false);  // pos
  k_qkp<<<ewb, 256, 0, st>>>(w.q, w.big0, w.big2, w.big3, BNKF);               // q - kf + pos
  gemm(st, w.big3, t.g1.w, t.g1.b, nullptr, w.big0, (int)BNK, DM, DM, true);   // relu(g1)
  gemm(st, w.big0, t.g2.w, t.g2.b, nullptr, w.big3, (int)BNK, DM, DM, false);  // attn
  int fwb = (int)((BNF + 255) / 256);
  k_softmaxk<<<fwb, 256, 0, st>>>(w.big3, BNF);
  k_attnred<<<fwb, 256, 0, st>>>(w.big3, w.big1, w.big2, w.res, BNF);
  gemm(st, w.res, t.fc2.w, t.fc2.b, h, h, BN, DM, din, false);                 // + residual
}

extern "C" void kernel_launch(void* const* d_in, const int* in_sizes, int n_in,
                              void* d_out, int out_size, void* d_ws, size_t ws_size,
                              hipStream_t stream)
{
  (void)out_size;
  if (n_in < 231) return;

  // --- detect flattening convention ---
  const float* xin;
  Cur cur; bool srt;
  if (in_sizes[0] == BB * NP * 6) {              // x first
    xin = (const float*)d_in[0];
    cur.d = d_in; cur.i = 1;
    srt = (in_sizes[1] == 32);                   // fc1a.b first => sorted keys
  } else {                                       // x last (fully sorted pytree)
    xin = (const float*)d_in[n_in - 1];
    cur.d = d_in; cur.i = 0;
    srt = true;
  }
  Net P; readNet(cur, srt, P);

  // --- workspace arena ---
  char* base = (char*)d_ws; size_t off = 0;
  auto aF = [&](size_t n) -> float* { float* p = (float*)(base + off); off = (off + n * 4 + 255) & ~(size_t)255; return p; };
  auto aI = [&](size_t n) -> int*   { int*   p = (int*)(base + off);   off = (off + n * 4 + 255) & ~(size_t)255; return p; };
  WS w;
  const size_t BIG  = (size_t)BB * NP * KNN * DM;
  const size_t BNF0 = (size_t)BB * NP * DM;
  w.big0 = aF(BIG); w.big1 = aF(BIG); w.big2 = aF(BIG); w.big3 = aF(BIG);
  w.x = aF(BNF0); w.q = aF(BNF0); w.kx = aF(BNF0); w.vx = aF(BNF0); w.res = aF(BNF0);
  w.hA = aF(BNF0); w.hB = aF(BNF0);
  w.gA = aF((size_t)8388608); w.gB = aF((size_t)8388608);
  w.rel = aF((size_t)BB * NP * KNN * 3);
  w.idx = aI((size_t)BB * NP * KNN);
  for (int i = 0; i < 5; ++i) {
    w.sxyz[i] = aF((size_t)BB * (NP >> i) * 3);
    w.sh[i]   = aF((size_t)BB * (NP >> i) * (32 << i));
  }
  w.fps = aI((size_t)BB * 1024);
  w.i3  = aI((size_t)BB * NP * 3);
  w.w3  = aF((size_t)BB * NP * 3);
  w.mean = aF(512); w.rstd = aF(512);
  if (off > ws_size) return;   // workspace too small: bail without corrupting memory

  hipStream_t st = stream;

  // --- stem ---
  k_slice3<<<cdiv(BB * NP * 3, 256), 256, 0, st>>>(xin, w.sxyz[0], BB * NP * 3);
  gemm(st, xin,  P.fc1a.w, P.fc1a.b, nullptr, w.hA,    BB * NP, 6,  32, true);
  gemm(st, w.hA, P.fc1b.w, P.fc1b.b, nullptr, w.sh[0], BB * NP, 32, 32, false);
  transformer(st, P.t1, w.sxyz[0], w.sh[0], NP, 32, w);

  // --- encoder ---
  int Ncur = NP, Cprev = 32;
  float* curh = w.sh[0]; float* curxyz = w.sxyz[0];
  for (int i = 0; i < 4; ++i) {
    int np = NP >> (i + 1), ch = 32 << (i + 1), cin = Cprev + 3;
    k_fps<<<BB, 256, 0, st>>>(curxyz, Ncur, np, w.fps);
    long t3 = (long)BB * np * 3;
    k_gather<<<(int)((t3 + 255) / 256), 256, 0, st>>>(curxyz, w.fps, w.sxyz[i+1], np, Ncur, 3, t3);
    k_knn16<<<cdiv(BB * np, 128), 128, 0, st>>>(w.sxyz[i+1], curxyz, np, Ncur, w.idx);
    long rows = (long)BB * np * KNN;
    long tc = rows * cin;
    k_tdcat<<<(int)((tc + 255) / 256), 256, 0, st>>>(curxyz, w.sxyz[i+1], curh, w.idx, w.gA, np, Ncur, Cprev, tc);
    gemm(st, w.gA, P.td0[i].w, P.td0[i].b, nullptr, w.gB, (int)rows, cin, ch, false);
    bnrelu(st, w.gB, P.td0[i], (int)rows, ch, w);
    gemm(st, w.gB, P.td1[i].w, P.td1[i].b, nullptr, w.gA, (int)rows, ch, ch, false);
    bnrelu(st, w.gA, P.td1[i], (int)rows, ch, w);
    long tm = (long)BB * np * ch;
    k_maxk<<<(int)((tm + 255) / 256), 256, 0, st>>>(w.gA, w.sh[i+1], ch, tm);
    transformer(st, P.tf[i], w.sxyz[i+1], w.sh[i+1], np, ch, w);
    Ncur = np; Cprev = ch; curh = w.sh[i+1]; curxyz = w.sxyz[i+1];
  }

  // --- bottleneck MLP + transformer ---
  gemm(st, curh, P.fc2a.w, P.fc2a.b, nullptr, w.hA, BB * Ncur, 512, 512, true);
  gemm(st, w.hA, P.fc2b.w, P.fc2b.b, nullptr, w.hB, BB * Ncur, 512, 512, true);
  gemm(st, w.hB, P.fc2c.w, P.fc2c.b, nullptr, w.hA, BB * Ncur, 512, 512, false);
  float* h = w.hA;
  transformer(st, P.t2, curxyz, h, Ncur, 512, w);

  // --- decoder ---
  for (int i = 0; i < 4; ++i) {
    int lvl = 3 - i, Nf = NP >> lvl, ch = 32 << lvl;
    gemm(st, h, P.tuf1[i].w, P.tuf1[i].b, nullptr, w.gA, BB * Ncur, 2 * ch, ch, false);
    bnrelu(st, w.gA, P.tuf1[i], BB * Ncur, ch, w);
    gemm(st, w.sh[lvl], P.tuf2[i].w, P.tuf2[i].b, nullptr, w.gB, BB * Nf, ch, ch, false);
    bnrelu(st, w.gB, P.tuf2[i], BB * Nf, ch, w);
    k_knn3w<<<cdiv(BB * Nf, 128), 128, 0, st>>>(w.sxyz[lvl], curxyz, Nf, Ncur, w.i3, w.w3);
    float* hout = (h == w.hA) ? w.hB : w.hA;
    long ti = (long)BB * Nf * ch;
    k_interp3<<<(int)((ti + 255) / 256), 256, 0, st>>>(w.gA, w.gB, w.i3, w.w3, hout, Nf, Ncur, ch, ti);
    h = hout; curxyz = w.sxyz[lvl]; Ncur = Nf;
    transformer(st, P.tutf[i], curxyz, h, Ncur, ch, w);
  }

  // --- head ---
  float* o1 = (h == w.hA) ? w.hB : w.hA;
  gemm(st, h,  P.fc3a.w, P.fc3a.b, nullptr, o1, BB * NP, 32, 64, true);
  gemm(st, o1, P.fc3b.w, P.fc3b.b, nullptr, h,  BB * NP, 64, 64, true);
  gemm(st, h,  P.fc3c.w, P.fc3c.b, nullptr, (float*)d_out, BB * NP, 64, 13, false);
}
